// QSwinBlock_27273042330108
// MI455X (gfx1250) — matrix operand order repaired and checked by reading, not compiled, not run on hardware
//
#include <hip/hip_runtime.h>

// ---------------------------------------------------------------------------
// Swin block for MI455X (gfx1250, wave32, WMMA).
// bf16 activations between stages, f32 WMMA accumulation, fused epilogues.
// ---------------------------------------------------------------------------

typedef __bf16 v16bf __attribute__((ext_vector_type(16)));
typedef __bf16 v8bf  __attribute__((ext_vector_type(8)));
typedef float  v8f   __attribute__((ext_vector_type(8)));

#define M_TOK 200704   // B * H * W = 64*56*56
#define C_DIM 128

// Window-partition row -> flat token index (shift = 3, 7x7 windows, 56x56, 64 imgs).
// Same mapping serves forward partition (gather) and reverse (scatter target).
__device__ __forceinline__ int win_remap(int r) {
  int wi = r / 49, p = r - wi * 49;
  int bb = wi >> 6, wl = wi & 63;
  int i = p / 7, j = p - i * 7;
  int h = (wl >> 3) * 7 + i + 3; if (h >= 56) h -= 56;
  int w = (wl & 7) * 7 + j + 3;  if (w >= 56) w -= 56;
  return bb * 3136 + h * 56 + w;
}

// ---------------------------------------------------------------------------
// Precompute rel-pos-bias + attention mask, padded to 64x64 per (window, head).
// addbuf[(win64*4 + head)][n*64 + m]; pads (n>=49 || m>=49) filled with -1e30.
// Only 64 windows x 4 heads exist -> 4 MB table, L2-resident, reused 64x.
// ---------------------------------------------------------------------------
__global__ __launch_bounds__(256) void bias_mask_kernel(
    const float* __restrict__ bt, const int* __restrict__ ri,
    const float* __restrict__ am, float* __restrict__ addbuf) {
  const int wimg = blockIdx.x >> 2;
  const int hd   = blockIdx.x & 3;
  float* dst = addbuf + (size_t)blockIdx.x * 4096;
  const float* mask = am + (size_t)wimg * 2401;
  for (int i = threadIdx.x; i < 4096; i += 256) {
    const int n = i >> 6, m = i & 63;
    float v = -1e30f;
    if (n < 49 && m < 49)
      v = bt[ri[n * 49 + m] * 4 + hd] + mask[n * 49 + m];
    dst[i] = v;
  }
}

// ---------------------------------------------------------------------------
// LayerNorm over C=128, one wave per token (lane holds 4 channels).
// windowed=1: output row r is LN of source token win_remap(r) (shift+partition).
// ---------------------------------------------------------------------------
__global__ __launch_bounds__(256) void ln_kernel(
    const float* __restrict__ x, const float* __restrict__ g,
    const float* __restrict__ bta, __bf16* __restrict__ out, int windowed) {
  const int row  = blockIdx.x * 8 + (threadIdx.x >> 5);
  const int lane = threadIdx.x & 31;
  const int src  = windowed ? win_remap(row) : row;

  const float4 v = ((const float4*)(x + (size_t)src * C_DIM))[lane];
  float s = v.x + v.y + v.z + v.w;
  #pragma unroll
  for (int o = 16; o > 0; o >>= 1) s += __shfl_xor(s, o, 32);
  const float mu = s * (1.0f / 128.0f);
  const float d0 = v.x - mu, d1 = v.y - mu, d2 = v.z - mu, d3 = v.w - mu;
  float q = d0 * d0 + d1 * d1 + d2 * d2 + d3 * d3;
  #pragma unroll
  for (int o = 16; o > 0; o >>= 1) q += __shfl_xor(q, o, 32);
  const float inv = rsqrtf(q * (1.0f / 128.0f) + 1e-5f);

  const float4 gw = ((const float4*)g)[lane];
  const float4 bw = ((const float4*)bta)[lane];
  __bf16* o4 = out + (size_t)row * C_DIM + lane * 4;
  o4[0] = (__bf16)(d0 * inv * gw.x + bw.x);
  o4[1] = (__bf16)(d1 * inv * gw.y + bw.y);
  o4[2] = (__bf16)(d2 * inv * gw.z + bw.z);
  o4[3] = (__bf16)(d3 * inv * gw.w + bw.w);
}

// ---------------------------------------------------------------------------
// Generic WMMA GEMM:  out[M,N] = A[M,K](bf16) * W[N,K](f32)^T + bias, epilogue.
// Block tile 128x64, 8 waves, wave w -> rows 16w..16w+15, 4 col tiles of 16.
// EP: 0=QKV(scale q cols, bf16 out)  1=PROJ(remap+residual, f32 out)
//     2=GELU(bf16 out)               3=FC2(residual, f32 out)
// ---------------------------------------------------------------------------
template <int EP>
__global__ __launch_bounds__(256) void gemm_kernel(
    const __bf16* __restrict__ A, const float* __restrict__ W,
    const float* __restrict__ bias, __bf16* __restrict__ outb,
    float* __restrict__ outf, const float* __restrict__ resid,
    int ldc, int kT) {
  __shared__ alignas(32) __bf16 As[128 * 128];
  __shared__ alignas(32) __bf16 Ws[64 * 128];
  const int tid  = threadIdx.x;
  const int wav  = tid >> 5;
  const int lane = tid & 31;
  const int l15  = lane & 15;
  const int half = lane >> 4;
  const int bm0  = blockIdx.x * 128;
  const int bn0  = blockIdx.y * 64;
  const int K    = kT * 128;

  v8f acc[4] = {};

  for (int kt = 0; kt < kT; ++kt) {
    const int k0 = kt * 128;
    // Stage A tile (already bf16): 128 rows x 128 k
    for (int c = tid; c < 128 * 16; c += 256) {
      const int r = c >> 4, c8 = (c & 15) << 3;
      *(v8bf*)&As[r * 128 + c8] =
          *(const v8bf*)&A[(size_t)(bm0 + r) * K + k0 + c8];
      if (kt + 1 < kT)  // global_prefetch for next K tile (FC2 path)
        __builtin_prefetch(&A[(size_t)(bm0 + r) * K + k0 + 128 + c8], 0, 1);
    }
    // Stage W tile, f32 -> bf16, layout Ws[n][k] (k contiguous)
    for (int c = tid; c < 64 * 32; c += 256) {
      const int nr = c >> 5, c4 = (c & 31) << 2;
      const float4 w4 = *(const float4*)&W[(size_t)(bn0 + nr) * K + k0 + c4];
      __bf16* dst = &Ws[nr * 128 + c4];
      dst[0] = (__bf16)w4.x; dst[1] = (__bf16)w4.y;
      dst[2] = (__bf16)w4.z; dst[3] = (__bf16)w4.w;
    }
    __syncthreads();
    #pragma unroll
    for (int ks = 0; ks < 4; ++ks) {
      // A fragment 16x32: lane half 0 -> K {0..15}, half 1 -> {16..31}
      const __bf16* ap = &As[(wav * 16 + l15) * 128 + ks * 32 + half * 16];
      const v8bf a0 = *(const v8bf*)ap;
      const v8bf a1 = *(const v8bf*)(ap + 8);
      v16bf aF;
      #pragma unroll
      for (int i = 0; i < 8; ++i) { aF[i] = a0[i]; aF[8 + i] = a1[i]; }
      #pragma unroll
      for (int ct = 0; ct < 4; ++ct) {
        // B fragment 32x16: lane -> col l15, K base half*16, contiguous in Ws
        const v16bf bF =
            *(const v16bf*)&Ws[(ct * 16 + l15) * 128 + ks * 32 + half * 16];
        acc[ct] = __builtin_amdgcn_wmma_f32_16x16x32_bf16(
            false, aF, false, bF, (short)0, acc[ct], false, false);
      }
    }
    __syncthreads();
  }

  // Epilogue: C frag lane layout -> row (half*8 + r), col l15
  const int gmb = bm0 + wav * 16 + half * 8;
  #pragma unroll
  for (int ct = 0; ct < 4; ++ct) {
    const int gc = bn0 + ct * 16 + l15;
    const float bv = bias[gc];
    #pragma unroll
    for (int r = 0; r < 8; ++r) {
      const int gm = gmb + r;
      float v = acc[ct][r] + bv;
      if (EP == 0) {              // QKV: scale q (first 128 cols) by HD^-0.5
        if (gc < 128) v *= 0.17677669529663687f;
        outb[(size_t)gm * ldc + gc] = (__bf16)v;
      } else if (EP == 1) {       // proj: window reverse + roll + residual
        const int t = win_remap(gm);
        outf[(size_t)t * C_DIM + gc] = resid[(size_t)t * C_DIM + gc] + v;
      } else if (EP == 2) {       // fc1: exact GELU
        v = 0.5f * v * (1.0f + erff(v * 0.70710678118654752f));
        outb[(size_t)gm * ldc + gc] = (__bf16)v;
      } else {                    // fc2: residual into final output
        outf[(size_t)gm * C_DIM + gc] = resid[(size_t)gm * C_DIM + gc] + v;
      }
    }
  }
}

// ---------------------------------------------------------------------------
// Windowed attention: one wave per (window, head). N=49 padded to 64.
// All K^T and V fragments are preloaded once (they are mt-invariant), so the
// global gathers issue early and overlap the S WMMAs and the softmax.
// S = q k^T with the bias+mask table fed through the WMMA C operand.
// Pad rows/cols carry -1e30 -> zero softmax weight; no divergence in epilogue.
// ---------------------------------------------------------------------------
__global__ __launch_bounds__(32) void attn_kernel(
    const __bf16* __restrict__ qkv, const float* __restrict__ addbuf,
    __bf16* __restrict__ attn_out) {
  __shared__ alignas(32) float sm[64 * 64];
  const int win  = blockIdx.x;
  const int hd   = blockIdx.y;
  const int lane = threadIdx.x;
  const int l15  = lane & 15;
  const int half = lane >> 4;

  const __bf16* base = qkv + (size_t)win * 49 * 384 + hd * 32;  // q; +128 k; +256 v
  const float* addp = addbuf + (size_t)((win & 63) * 4 + hd) * 4096;

  // ---- preload K^T B-fragments for all 4 column tiles (mt-invariant) ----
  v16bf bF[4];
  #pragma unroll
  for (int nt = 0; nt < 4; ++nt) {
    const int bcol = nt * 16 + l15;
    v16bf t = {};
    if (bcol < 49)  // B[k][n] = k[n][k]: contiguous in head-dim
      t = *(const v16bf*)(base + (size_t)bcol * 384 + 128 + half * 16);
    bF[nt] = t;
  }

  // ---- preload V B-fragments for all (ks, ct) tiles (mt-invariant) ----
  const __bf16* vb = base + 256;
  v16bf vF[4];
  #pragma unroll
  for (int ks = 0; ks < 2; ++ks) {
    #pragma unroll
    for (int ct = 0; ct < 2; ++ct) {
      const int d = ct * 16 + l15;
      const int kb = ks * 32 + half * 16;
      v16bf t = {};
      #pragma unroll
      for (int j = 0; j < 16; ++j) {
        const int tok = kb + j;
        t[j] = (tok < 49) ? vb[(size_t)tok * 384 + d] : (__bf16)0.0f;
      }
      vF[ks * 2 + ct] = t;
    }
  }

  // ---- S = q @ k^T : 4x4 tiles of 16x16, single K step of 32, C = bias+mask ----
  for (int mt = 0; mt < 4; ++mt) {
    const int arow = mt * 16 + l15;
    v16bf aF = {};
    if (arow < 49) {
      const __bf16* ap = base + (size_t)arow * 384 + half * 16;
      const v8bf a0 = *(const v8bf*)ap;
      const v8bf a1 = *(const v8bf*)(ap + 8);
      #pragma unroll
      for (int i = 0; i < 8; ++i) { aF[i] = a0[i]; aF[8 + i] = a1[i]; }
    }
    #pragma unroll
    for (int nt = 0; nt < 4; ++nt) {
      const int cbase = (mt * 16 + half * 8) * 64 + nt * 16 + l15;
      v8f c;
      #pragma unroll
      for (int r = 0; r < 8; ++r) c[r] = addp[cbase + r * 64];
      c = __builtin_amdgcn_wmma_f32_16x16x32_bf16(
          false, aF, false, bF[nt], (short)0, c, false, false);
      #pragma unroll
      for (int r = 0; r < 8; ++r) sm[cbase + r * 64] = c[r];
    }
  }
  __syncthreads();

  // ---- row softmax in LDS (uniform over all 64 rows; pad rows are junk) ----
  for (int row = lane; row < 64; row += 32) {
    float mx = -1e30f;
    for (int c2 = 0; c2 < 64; ++c2) mx = fmaxf(mx, sm[row * 64 + c2]);
    float s = 0.0f;
    for (int c2 = 0; c2 < 64; ++c2) {
      const float e = __expf(sm[row * 64 + c2] - mx);
      sm[row * 64 + c2] = e; s += e;
    }
    const float inv = 1.0f / s;
    for (int c2 = 0; c2 < 64; ++c2) sm[row * 64 + c2] *= inv;
  }
  __syncthreads();

  // ---- O = P @ V : 4 row tiles x 2 col tiles, K = 64 (2 steps) ----
  for (int mt = 0; mt < 4; ++mt) {
    v8f acc0 = {}, acc1 = {};
    #pragma unroll
    for (int ks = 0; ks < 2; ++ks) {
      const float* pp = &sm[(mt * 16 + l15) * 64 + ks * 32 + half * 16];
      v16bf pF;
      #pragma unroll
      for (int i = 0; i < 8; ++i) {
        pF[i] = (__bf16)pp[i]; pF[8 + i] = (__bf16)pp[8 + i];
      }
      acc0 = __builtin_amdgcn_wmma_f32_16x16x32_bf16(
          false, pF, false, vF[ks * 2 + 0], (short)0, acc0, false, false);
      acc1 = __builtin_amdgcn_wmma_f32_16x16x32_bf16(
          false, pF, false, vF[ks * 2 + 1], (short)0, acc1, false, false);
    }
    #pragma unroll
    for (int r = 0; r < 8; ++r) {
      const int row = mt * 16 + half * 8 + r;
      if (row < 49) {
        const size_t o = ((size_t)win * 49 + row) * C_DIM + hd * 32;
        attn_out[o + l15]      = (__bf16)acc0[r];
        attn_out[o + 16 + l15] = (__bf16)acc1[r];
      }
    }
  }
}

// ---------------------------------------------------------------------------
extern "C" void kernel_launch(void* const* d_in, const int* in_sizes, int n_in,
                              void* d_out, int out_size, void* d_ws,
                              size_t ws_size, hipStream_t stream) {
  const float* x      = (const float*)d_in[0];
  const float* n1w    = (const float*)d_in[1];
  const float* n1b    = (const float*)d_in[2];
  const float* qkv_w  = (const float*)d_in[3];
  const float* qkv_b  = (const float*)d_in[4];
  const float* proj_w = (const float*)d_in[5];
  const float* proj_b = (const float*)d_in[6];
  const float* n2w    = (const float*)d_in[7];
  const float* n2b    = (const float*)d_in[8];
  const float* fc1_w  = (const float*)d_in[9];
  const float* fc1_b  = (const float*)d_in[10];
  const float* fc2_w  = (const float*)d_in[11];
  const float* fc2_b  = (const float*)d_in[12];
  const float* btab   = (const float*)d_in[13];
  const float* amask  = (const float*)d_in[14];
  const int*   ridx   = (const int*)d_in[15];
  float* out = (float*)d_out;

  const size_t M = M_TOK;
  char* w = (char*)d_ws;
  // [0, M*1024)            : qkv bf16 (M*384), later gelu bf16 (M*512)
  // [M*1024, +M*256)       : LN output bf16 (A operand for QKV / FC1)
  // [.., +M*256)           : attention output bf16 (A operand for proj)
  // [.., +M*512)           : x1 f32 (post-attn residual stream)
  // [M*2048, +4MB)         : padded bias+mask table [64 win][4 head][64*64] f32
  __bf16* big    = (__bf16*)w;
  __bf16* abuf   = (__bf16*)(w + M * 1024);
  __bf16* aobuf  = (__bf16*)(w + M * 1024 + M * 256);
  float*  x1     = (float*)(w + M * 1024 + M * 512);
  float*  addbuf = (float*)(w + M * 2048);

  // 0) bias+mask table (64 windows x 4 heads, padded 64x64)
  bias_mask_kernel<<<dim3(256), 256, 0, stream>>>(btab, ridx, amask, addbuf);
  // 1) LN1 + shift + window partition -> bf16
  ln_kernel<<<dim3((unsigned)(M / 8)), 256, 0, stream>>>(x, n1w, n1b, abuf, 1);
  // 2) QKV GEMM (N=384), q scaled by HD^-0.5
  gemm_kernel<0><<<dim3(1568, 6), 256, 0, stream>>>(
      abuf, qkv_w, qkv_b, big, nullptr, nullptr, 384, 1);
  // 3) Windowed attention (4096 windows x 4 heads)
  attn_kernel<<<dim3(4096, 4), 32, 0, stream>>>(big, addbuf, aobuf);
  // 4) proj GEMM + window reverse + roll + residual -> x1 (f32)
  gemm_kernel<1><<<dim3(1568, 2), 256, 0, stream>>>(
      aobuf, proj_w, proj_b, nullptr, x1, x, 128, 1);
  // 5) LN2 -> bf16
  ln_kernel<<<dim3((unsigned)(M / 8)), 256, 0, stream>>>(x1, n2w, n2b, abuf, 0);
  // 6) FC1 GEMM (N=512) + exact GELU -> bf16
  gemm_kernel<2><<<dim3(1568, 8), 256, 0, stream>>>(
      abuf, fc1_w, fc1_b, big, nullptr, nullptr, 512, 1);
  // 7) FC2 GEMM (K=512, 4 k-tiles) + residual -> d_out
  gemm_kernel<3><<<dim3(1568, 2), 256, 0, stream>>>(
      big, fc2_w, fc2_b, nullptr, out, x1, 128, 4);
}
